// TrxGNNGPT_5136780886106
// MI455X (gfx1250) — matrix operand order repaired and checked by reading, not compiled, not run on hardware
//
#include <hip/hip_runtime.h>
#include <hip/hip_bf16.h>

// ---------------- problem constants (from reference) ----------------
#define VOCAB   10000
#define D       64
#define L       8
#define DG      256
#define NH      4
#define HD      64          // DG/NH
#define DFF     1024        // 4*DG
#define NNODES  8192
#define NEDGES  2048
#define FD      512         // L*D
#define NTOK    (NEDGES*L)  // 16384 transformer rows

typedef _Float16 hf;
typedef __attribute__((ext_vector_type(16))) _Float16 v16h;
typedef __attribute__((ext_vector_type(8)))  _Float16 v8h;
typedef __attribute__((ext_vector_type(8)))  float    v8f;

// ======================================================================
// WMMA GEMM:  C[M,N] = act( A[M,K](f32) * B[K,N](f16) + bias + addend )
//
//   Block = 4 waves stacked along M; block tile 128 x 32.
//   Per wave: 32x32 output = 2 (M-subtiles) x 2 (N-subtiles) 16x16 WMMA
//   accumulators -> 4 v_wmma per K-step, A fragments reused across N,
//   B fragments reused across M.
//   B tile [32k x 32n] staged through double-buffered LDS (f16, padded
//   rows, one barrier per iteration); next tile's global load issued
//   before the current WMMAs so global latency hides under matrix math.
//   Epilogue is compile-time specialized:
//     ACT:     0 none, 1 relu, 2 tanh-gelu
//     ADDMODE: 0 none, 1 addend[m], 2 addend[gather[m]]
//     BIAS:    bias[n] or not
//   Ragged N (lm_head N=10000): block-uniform subtile guards -> EXEC all-1s
//   at every WMMA, no divergent __syncthreads.
// ======================================================================
__device__ __forceinline__ v16h load_a_frag(const float* __restrict__ Arow,
                                            int half)
{
    v16h a;
#pragma unroll
    for (int v = 0; v < 8; ++v) {
        const int k0 = ((v & 3) << 1) + ((v >> 2) << 4) + (half << 3);
        const float2 av = *(const float2*)(Arow + k0);
        a[2 * v]     = (hf)av.x;
        a[2 * v + 1] = (hf)av.y;
    }
    return a;
}

template<int ACT, int ADDMODE, bool BIAS>
__global__ __launch_bounds__(128)
void gemm_wmma_kernel(const float* __restrict__ A, const hf* __restrict__ B,
                      float* __restrict__ C,
                      const float* __restrict__ bias,
                      const float* __restrict__ addend,
                      const int*   __restrict__ gather,
                      int M, int N, int K)
{
    __shared__ hf bsh[2][32][40];        // ping-pong 32k x 32n, 80B rows

    const int tid  = threadIdx.x;
    const int wave = tid >> 5;
    const int lane = tid & 31;
    const int half = lane >> 4;
    const int mA   = lane & 15;
    const int ntile = blockIdx.x * 32;
    const int m0    = (blockIdx.y * 4 + wave) * 32;
    const bool n1ok = (ntile + 16) < N;  // block-uniform

    // cooperative B-tile load: 128 thr x 8 f16 = 32x32 tile
    const int brow = tid >> 2;           // k within tile
    const int bcol = (tid & 3) * 8;      // n within tile
    const bool bvalid = (ntile + bcol + 8) <= N;

    const int klB = (lane & 15) + (half << 4);   // B-fragment k row

    v8f acc00 = {}, acc01 = {}, acc10 = {}, acc11 = {};

    const int nIter = K >> 5;
    // preload tile 0
    float4 braw = {0.f, 0.f, 0.f, 0.f};
    if (bvalid)
        braw = *(const float4*)(B + (size_t)brow * N + ntile + bcol);
    *(float4*)&bsh[0][brow][bcol] = braw;

    for (int it = 0; it < nIter; ++it) {
        const int p  = it & 1;
        const int kk = it << 5;

        // issue next tile's global load early (hides under WMMAs below)
        if (it + 1 < nIter) {
            braw = make_float4(0.f, 0.f, 0.f, 0.f);
            if (bvalid)
                braw = *(const float4*)(B + (size_t)(kk + 32 + brow) * N
                                          + ntile + bcol);
        }
        __syncthreads();                 // bsh[p] writes visible; prior reads done

        // A fragments (f32 -> f16) for the two M-subtiles
        const v16h a0 = load_a_frag(A + (size_t)(m0 + mA) * K + kk, half);
        const v16h a1 = load_a_frag(A + (size_t)(m0 + 16 + mA) * K + kk, half);

        // B fragments from LDS (lane = n column, 16 k each in table order)
        v16h b0, b1;
#pragma unroll
        for (int i = 0; i < 16; ++i) {
            const int kb = (i & 7) + ((i >> 3) << 4) + (half << 3);
            b0[i] = bsh[p][kb][mA];
            b1[i] = bsh[p][kb][16 + mA];
        }

        acc00 = __builtin_amdgcn_wmma_f32_16x16x32_f16(
                    false, a0, false, b0, (short)0, acc00, false, false);
        acc10 = __builtin_amdgcn_wmma_f32_16x16x32_f16(
                    false, a1, false, b0, (short)0, acc10, false, false);
        if (n1ok) {
            acc01 = __builtin_amdgcn_wmma_f32_16x16x32_f16(
                        false, a0, false, b1, (short)0, acc01, false, false);
            acc11 = __builtin_amdgcn_wmma_f32_16x16x32_f16(
                        false, a1, false, b1, (short)0, acc11, false, false);
        }

        // fill the other buffer for next iteration
        if (it + 1 < nIter)
            *(float4*)&bsh[p ^ 1][brow][bcol] = braw;
    }

    // ---- straight-line templated epilogue ----
    const int nC = lane & 15;
    auto epi = [&](float r, int m, int n) -> float {
        if (BIAS)         r += bias[n];
        if (ADDMODE == 1) r += addend[(size_t)m * N + n];
        if (ADDMODE == 2) r += addend[(size_t)gather[m] * N + n];
        if (ACT == 1)     r = fmaxf(r, 0.f);
        if (ACT == 2) {
            const float t = 0.7978845608f * (r + 0.044715f * r * r * r);
            r = 0.5f * r * (1.f + tanhf(t));
        }
        return r;
    };
#pragma unroll
    for (int v = 0; v < 8; ++v) {
        const int mlo = m0 + v + (half << 3);        // M-subtile 0 row
        const int mhi = mlo + 16;                    // M-subtile 1 row
        const int n0  = ntile + nC;
        C[(size_t)mlo * N + n0] = epi(acc00[v], mlo, n0);
        C[(size_t)mhi * N + n0] = epi(acc10[v], mhi, n0);
        if (n1ok) {
            const int n1 = n0 + 16;
            C[(size_t)mlo * N + n1] = epi(acc01[v], mlo, n1);
            C[(size_t)mhi * N + n1] = epi(acc11[v], mhi, n1);
        }
    }
}

// ---------------- weight prep: f32 -> f16 (once per launch) ----------------
__global__ void f32_to_f16_kernel(const float* __restrict__ src,
                                  hf* __restrict__ dst, int n)
{
    const int i = blockIdx.x * blockDim.x + threadIdx.x;
    if (i < n) dst[i] = (hf)src[i];
}

// embT[d, v] = emb[v, d]  (tied lm-head weight, f16, row-major [D, VOCAB])
__global__ void transpose_emb_f16_kernel(const float* __restrict__ emb,
                                         hf* __restrict__ embT)
{
    const int i = blockIdx.x * blockDim.x + threadIdx.x;
    if (i >= D * VOCAB) return;
    const int v = i % VOCAB, d = i / VOCAB;
    embT[i] = (hf)emb[(size_t)v * D + d];
}

// ---------------- MLM masking: labels + masked tokens ----------------
__global__ void mask_kernel(const int* __restrict__ etok,
                            const unsigned char* __restrict__ mrow,
                            float* __restrict__ labels,
                            int* __restrict__ mtok)
{
    const int i = blockIdx.x * blockDim.x + threadIdx.x;
    if (i >= NTOK) return;
    const int e = i / L;
    const int t = etok[i];
    const bool special = (t >= 0) && (t < 4);          // SPECIAL = {0,1,2,3}
    const bool m = (mrow[e] != 0) && !special;
    labels[i] = m ? (float)t : -100.0f;
    mtok[i]   = m ? 4 : t;                              // MASK_ID = 4
}

// ---------------- embedding gather: out[r, l*64+d] = emb[tok[r,l], d] ----
__global__ void gather_embed_kernel(const int* __restrict__ tok,
                                    const float* __restrict__ emb,
                                    float* __restrict__ out, int rows)
{
    const int i = blockIdx.x * blockDim.x + threadIdx.x;
    if (i >= rows * L * D) return;
    const int d = i & (D - 1);
    const int tl = i >> 6;
    out[i] = emb[(size_t)tok[tl] * D + d];
}

__global__ void fill_zero_kernel(float* __restrict__ p, int n)
{
    const int i = blockIdx.x * blockDim.x + threadIdx.x;
    if (i < n) p[i] = 0.f;
}

// ---------------- segment-sum: agg[dst[e]] += msg[e] ----------------
__global__ void scatter_add_kernel(const float* __restrict__ msg,
                                   const int* __restrict__ dst,
                                   float* __restrict__ agg)
{
    const int i = blockIdx.x * blockDim.x + threadIdx.x;
    if (i >= NEDGES * FD) return;
    const int e = i >> 9, c = i & (FD - 1);
    atomicAdd(&agg[(size_t)dst[e] * FD + c], msg[i]);
}

// ---------------- eh[e] = h[src[e]] + h[dst[e]] ----------------
__global__ void eh_kernel(const float* __restrict__ h,
                          const int* __restrict__ src,
                          const int* __restrict__ dst,
                          float* __restrict__ eh)
{
    const int i = blockIdx.x * blockDim.x + threadIdx.x;
    if (i >= NEDGES * FD) return;
    const int e = i >> 9, c = i & (FD - 1);
    eh[i] = h[(size_t)src[e] * FD + c] + h[(size_t)dst[e] * FD + c];
}

// ---------------- per-edge MHA: S=8, 4 heads of 64 (VALU; tiles far
// below 16x16 WMMA granularity). One block per edge. ----------------
__global__ __launch_bounds__(DG)
void attention_kernel(const float* __restrict__ Q, const float* __restrict__ Kc,
                      const float* __restrict__ Vc, float* __restrict__ O)
{
    __shared__ float qs[L][DG], ks[L][DG], vs[L][DG];
    __shared__ float aw[NH][L][L];
    const int b = blockIdx.x, t = threadIdx.x;
    const size_t base = (size_t)b * L * DG;
#pragma unroll
    for (int r = 0; r < L; ++r) {
        qs[r][t] = Q[base + r * DG + t];
        ks[r][t] = Kc[base + r * DG + t];
        vs[r][t] = Vc[base + r * DG + t];
    }
    __syncthreads();
    if (t < NH * L) {                    // 32 threads: one (head, query) each
        const int h = t >> 3, qi = t & 7;
        float s[L];
        float mx = -3.4e38f;
#pragma unroll
        for (int ki = 0; ki < L; ++ki) {
            float acc = 0.f;
            for (int d = 0; d < HD; ++d)
                acc += qs[qi][h * HD + d] * ks[ki][h * HD + d];
            s[ki] = acc * 0.125f;        // 1/sqrt(64)
            mx = fmaxf(mx, s[ki]);
        }
        float sum = 0.f;
#pragma unroll
        for (int ki = 0; ki < L; ++ki) { s[ki] = __expf(s[ki] - mx); sum += s[ki]; }
        const float inv = 1.f / sum;
#pragma unroll
        for (int ki = 0; ki < L; ++ki) aw[h][qi][ki] = s[ki] * inv;
    }
    __syncthreads();
    const int h = t >> 6, d = t & (HD - 1);
#pragma unroll
    for (int qi = 0; qi < L; ++qi) {
        float acc = 0.f;
#pragma unroll
        for (int ki = 0; ki < L; ++ki)
            acc += aw[h][qi][ki] * vs[ki][h * HD + d];
        O[base + qi * DG + h * HD + d] = acc;
    }
}

// ---------------- in-place row softmax over N columns ----------------
__global__ __launch_bounds__(256)
void softmax_rows_kernel(float* __restrict__ data, int N)
{
    __shared__ float red[256];
    float* row = data + (size_t)blockIdx.x * N;
    const int t = threadIdx.x;
    float mx = -3.4e38f;
    for (int j = t; j < N; j += 256) mx = fmaxf(mx, row[j]);
    red[t] = mx; __syncthreads();
    for (int s = 128; s > 0; s >>= 1) {
        if (t < s) red[t] = fmaxf(red[t], red[t + s]);
        __syncthreads();
    }
    mx = red[0]; __syncthreads();
    float sum = 0.f;
    for (int j = t; j < N; j += 256) {
        const float v = __expf(row[j] - mx);
        row[j] = v; sum += v;
    }
    red[t] = sum; __syncthreads();
    for (int s = 128; s > 0; s >>= 1) {
        if (t < s) red[t] += red[t + s];
        __syncthreads();
    }
    const float inv = 1.f / red[0];
    for (int j = t; j < N; j += 256) row[j] *= inv;
}

// ======================================================================
// launch
// ======================================================================
template<int ACT, int ADDMODE, bool BIAS>
static inline void launch_gemm(hipStream_t s, const float* A, const hf* B,
                               float* C, const float* bias, const float* addend,
                               const int* gather, int M, int N, int K)
{
    dim3 grid((N + 31) / 32, M / 128), block(128);
    hipLaunchKernelGGL((gemm_wmma_kernel<ACT, ADDMODE, BIAS>), grid, block,
                       0, s, A, B, C, bias, addend, gather, M, N, K);
}

static inline void conv16(hipStream_t s, const float* src, hf* dst, int n)
{
    hipLaunchKernelGGL(f32_to_f16_kernel, dim3((n + 255) / 256), dim3(256),
                       0, s, src, dst, n);
}

extern "C" void kernel_launch(void* const* d_in, const int* in_sizes, int n_in,
                              void* d_out, int out_size, void* d_ws, size_t ws_size,
                              hipStream_t stream)
{
    const int*   node_tok = (const int*)d_in[0];
    const int*   edge_tok = (const int*)d_in[1];
    const int*   eidx     = (const int*)d_in[2];          // [2, NEDGES]
    const unsigned char* mrow = (const unsigned char*)d_in[3];
    const float* emb   = (const float*)d_in[4];
    const float* W_msg = (const float*)d_in[5];
    const float* W_node= (const float*)d_in[6];
    const float* lc1_w = (const float*)d_in[7];
    const float* lc1_b = (const float*)d_in[8];
    const float* lc2_w = (const float*)d_in[9];
    const float* lc2_b = (const float*)d_in[10];
    const float* Wq = (const float*)d_in[11];
    const float* Wk = (const float*)d_in[12];
    const float* Wv = (const float*)d_in[13];
    const float* Wo = (const float*)d_in[14];
    const float* Wf1 = (const float*)d_in[15];
    const float* Wf2 = (const float*)d_in[16];

    const int* src = eidx;
    const int* dst = eidx + NEDGES;

    float* out_labels = (float*)d_out;                     // [NEDGES*L]
    float* out_probs  = (float*)d_out + NTOK;              // [NTOK, VOCAB]

    // ------- workspace layout (MiB offsets, lifetime-based reuse) -------
    char* ws = (char*)d_ws;
    #define WSF(mb) ((float*)(ws + ((size_t)(mb) << 20)))
    int*   mtok = (int*)ws;                                //  64 KB @ 0
    float* x    = WSF(1);    // [NNODES, FD]  16 MB  [1,17)
    float* e    = WSF(17);   // [NEDGES, FD]   4 MB
    float* msg  = WSF(21);   // [NEDGES, FD]   4 MB
    float* agg  = WSF(25);   // [NNODES, FD]  16 MB
    float* h    = WSF(41);   // [NNODES, FD]  16 MB
    float* eh   = WSF(1);    // [NEDGES, FD]   4 MB  (reuses x)
    float* z    = WSF(5);    // [NTOK, DG]    16 MB
    float* q    = WSF(21);   // [NTOK, DG]    16 MB  (reuses msg/agg)
    float* k    = WSF(41);   // [NTOK, DG]    16 MB  (reuses h)
    float* v    = WSF(57);   // [NTOK, DG]    16 MB
    float* o    = WSF(73);   // [NTOK, DG]    16 MB
    float* x1   = WSF(89);   // [NTOK, DG]    16 MB
    float* mid  = WSF(1);    // [NTOK, DFF]   64 MB  (reuses dead regions)
    float* x2   = WSF(105);  // [NTOK, DG]    16 MB
    float* y    = WSF(1);    // [NTOK, D]      4 MB  (reuses mid)

    // f16 weight pool @ 121 MiB (~4 MB total)
    size_t cur = (size_t)121 << 20;
    auto allocH = [&](size_t elems) {
        hf* p = (hf*)(ws + cur);
        cur += (elems * sizeof(hf) + 255) & ~(size_t)255;
        return p;
    };
    hf* Wmsg_h  = allocH(FD * FD);
    hf* Wnode_h = allocH(FD * FD);
    hf* lc1_h   = allocH(D * DG);
    hf* Wq_h    = allocH(DG * DG);
    hf* Wk_h    = allocH(DG * DG);
    hf* Wv_h    = allocH(DG * DG);
    hf* Wo_h    = allocH(DG * DG);
    hf* Wf1_h   = allocH(DG * DFF);
    hf* Wf2_h   = allocH(DFF * DG);
    hf* lc2_h   = allocH(DG * D);
    hf* embT_h  = allocH(D * VOCAB);

    const int B256 = 256;
    // 0) weight prep: f32 -> f16 copies + transposed tied lm-head weight
    conv16(stream, W_msg,  Wmsg_h,  FD * FD);
    conv16(stream, W_node, Wnode_h, FD * FD);
    conv16(stream, lc1_w,  lc1_h,   D * DG);
    conv16(stream, Wq,     Wq_h,    DG * DG);
    conv16(stream, Wk,     Wk_h,    DG * DG);
    conv16(stream, Wv,     Wv_h,    DG * DG);
    conv16(stream, Wo,     Wo_h,    DG * DG);
    conv16(stream, Wf1,    Wf1_h,   DG * DFF);
    conv16(stream, Wf2,    Wf2_h,   DFF * DG);
    conv16(stream, lc2_w,  lc2_h,   DG * D);
    hipLaunchKernelGGL(transpose_emb_f16_kernel,
                       dim3((D * VOCAB + B256 - 1) / B256), dim3(B256),
                       0, stream, emb, embT_h);

    // 1) MLM masking -> labels (d_out) + masked tokens
    hipLaunchKernelGGL(mask_kernel, dim3((NTOK + B256 - 1) / B256), dim3(B256),
                       0, stream, edge_tok, mrow, out_labels, mtok);
    // 2) embedding gathers
    hipLaunchKernelGGL(gather_embed_kernel,
                       dim3((NNODES * L * D + B256 - 1) / B256), dim3(B256),
                       0, stream, node_tok, emb, x, NNODES);
    hipLaunchKernelGGL(gather_embed_kernel,
                       dim3((NEDGES * L * D + B256 - 1) / B256), dim3(B256),
                       0, stream, mtok, emb, e, NEDGES);
    // 3) agg = 0
    hipLaunchKernelGGL(fill_zero_kernel,
                       dim3((NNODES * FD + B256 - 1) / B256), dim3(B256),
                       0, stream, agg, NNODES * FD);
    // 4) msg = relu(e @ W_msg + x[src])
    launch_gemm<1, 2, false>(stream, e, Wmsg_h, msg, nullptr, x, src,
                             NEDGES, FD, FD);
    // 5) agg = segment_sum(msg, dst)
    hipLaunchKernelGGL(scatter_add_kernel,
                       dim3((NEDGES * FD + B256 - 1) / B256), dim3(B256),
                       0, stream, msg, dst, agg);
    // 6) h = relu(x @ W_node + agg)
    launch_gemm<1, 1, false>(stream, x, Wnode_h, h, nullptr, agg, nullptr,
                             NNODES, FD, FD);
    // 7) eh = h[src] + h[dst]
    hipLaunchKernelGGL(eh_kernel,
                       dim3((NEDGES * FD + B256 - 1) / B256), dim3(B256),
                       0, stream, h, src, dst, eh);
    // 8) z = eh(view [NTOK, D]) @ lc1_w + lc1_b
    launch_gemm<0, 0, true>(stream, eh, lc1_h, z, lc1_b, nullptr, nullptr,
                            NTOK, DG, D);
    // 9) Q/K/V projections
    launch_gemm<0, 0, false>(stream, z, Wq_h, q, nullptr, nullptr, nullptr,
                             NTOK, DG, DG);
    launch_gemm<0, 0, false>(stream, z, Wk_h, k, nullptr, nullptr, nullptr,
                             NTOK, DG, DG);
    launch_gemm<0, 0, false>(stream, z, Wv_h, v, nullptr, nullptr, nullptr,
                             NTOK, DG, DG);
    // 10) per-edge attention
    hipLaunchKernelGGL(attention_kernel, dim3(NEDGES), dim3(DG), 0, stream,
                       q, k, v, o);
    // 11) x1 = z + o @ Wo
    launch_gemm<0, 1, false>(stream, o, Wo_h, x1, nullptr, z, nullptr,
                             NTOK, DG, DG);
    // 12) mid = gelu(x1 @ Wf1)
    launch_gemm<2, 0, false>(stream, x1, Wf1_h, mid, nullptr, nullptr, nullptr,
                             NTOK, DFF, DG);
    // 13) x2 = x1 + mid @ Wf2
    launch_gemm<0, 1, false>(stream, mid, Wf2_h, x2, nullptr, x1, nullptr,
                             NTOK, DG, DFF);
    // 14) y = x2 @ lc2_w + lc2_b
    launch_gemm<0, 0, true>(stream, x2, lc2_h, y, lc2_b, nullptr, nullptr,
                            NTOK, D, DG);
    // 15) logits = y @ embT  (row-major f16 weight; straight into d_out)
    launch_gemm<0, 0, false>(stream, y, embT_h, out_probs, nullptr, nullptr,
                             nullptr, NTOK, VOCAB, D);
    // 16) probs = softmax(logits) in place
    hipLaunchKernelGGL(softmax_rows_kernel, dim3(NTOK), dim3(256), 0, stream,
                       out_probs, VOCAB);
}
